// MultiHeadSelfAttention_65652870087334
// MI455X (gfx1250) — matrix operand order repaired and checked by reading, not compiled, not run on hardware
//
#include <hip/hip_runtime.h>

// ---------------------------------------------------------------------------
// MI455X (gfx1250, wave32) fused MHSA-style pipeline:
//   k1: qkv = x@W1+b1   (only Q,V columns; K unused by the reference)
//   k2: flash-style  softmax(Q@V^T * 1/8) @ V   per (b,head)
//   k3: out = attn@W2+b2
// v_wmma_f32_16x16x32_bf16 everywhere; bf16 intermediates in d_ws (~38MB,
// L2-resident). Async global->LDS (ASYNCcnt) for bf16 copy-through stages.
// ---------------------------------------------------------------------------

typedef __bf16 bf16;
typedef __attribute__((ext_vector_type(16))) __bf16 v16bf;
typedef __attribute__((ext_vector_type(8)))  __bf16 v8bf;
typedef __attribute__((ext_vector_type(8)))  float  v8f;

#define D_EMB 768
#define SEQ   2048
#define NB    4
#define NH    12
#define HD    64

__device__ __forceinline__ v8f wmma_bf16(v16bf a, v16bf b, v8f c) {
  return __builtin_amdgcn_wmma_f32_16x16x32_bf16(false, a, false, b, (short)0, c, false, false);
}

// A-matrix 16x32 bf16 fragment: lane L holds row M=L&15;
// lanes<16 get K {0..15}+kk, lanes>=16 get K {16..31}+kk.
__device__ __forceinline__ v16bf afrag(const bf16* row, int kk, int hi) {
  union { v16bf v; v8bf h[2]; } u;
  u.h[0] = *(const v8bf*)(row + kk + hi * 16);
  u.h[1] = *(const v8bf*)(row + kk + hi * 16 + 8);
  return u.v;
}

// 16-byte async global->LDS copy (GLOBAL_LOAD_ASYNC_TO_LDS_B128, ASYNCcnt).
// INST_OFFSET is added to BOTH the LDS and global addresses (ISA 10.2/15.18).
template <int OFF>
__device__ __forceinline__ void async_ld16(unsigned lds, unsigned long long g) {
  asm volatile("global_load_async_to_lds_b128 %0, %1, off offset:%2"
               :: "v"(lds), "v"(g), "i"(OFF) : "memory");
}
__device__ __forceinline__ void wait_async0() {
  asm volatile("s_wait_asynccnt 0x0" ::: "memory");
}
__device__ __forceinline__ unsigned lds_off(const void* p) {
  // low 32 bits of a flat shared-aperture address are the LDS byte offset
  return (unsigned)(unsigned long long)p;
}

// ---------------------------------------------------------------------------
// Kernel 1: qkv projection. grid=(M/128, NH): block y = one head's Q+V block
// (128 contiguous W1 columns). 256 thr = 8 waves; tile 128(M) x 128(N).
// ---------------------------------------------------------------------------
__global__ __launch_bounds__(256) void qkv_gemm(
    const float* __restrict__ x, const float* __restrict__ W1,
    const float* __restrict__ b1, bf16* __restrict__ Qb, bf16* __restrict__ Vb) {
  __shared__ __align__(32) bf16 Al[128 * 64];   // A tile [m][k] bf16
  __shared__ __align__(32) bf16 Bt[128 * 64];   // B tile transposed [n][k]

  const int t = threadIdx.x;
  const int wid = t >> 5, lane = t & 31, hi = lane >> 4, ln = lane & 15;
  const int mbase = blockIdx.x * 128;
  const int hd = blockIdx.y;
  const int cb = hd * 192 + 64;                 // Q cols then V cols, contiguous

  const int arow = t >> 1, ahalf = t & 1;       // A staging: 32 f32/thread
  const int bk = t >> 2, bq = t & 3;            // B staging: 32 f32/thread

  v8f acc[8] = {};

  for (int k0 = 0; k0 < D_EMB; k0 += 64) {
    {   // stage A: x (f32) -> bf16 LDS
      const float* src = x + (size_t)(mbase + arow) * D_EMB + k0 + ahalf * 32;
      bf16* dst = Al + arow * 64 + ahalf * 32;
      #pragma unroll
      for (int j = 0; j < 32; j += 4) {
        float4 f = *(const float4*)(src + j);
        dst[j + 0] = (bf16)f.x; dst[j + 1] = (bf16)f.y;
        dst[j + 2] = (bf16)f.z; dst[j + 3] = (bf16)f.w;
      }
    }
    {   // stage B transposed: W1[k0+k][cb+n] -> Bt[n][k]
      const float* src = W1 + (size_t)(k0 + bk) * (3 * D_EMB) + cb + bq * 32;
      #pragma unroll
      for (int j = 0; j < 32; j += 4) {
        float4 f = *(const float4*)(src + j);
        Bt[(bq * 32 + j + 0) * 64 + bk] = (bf16)f.x;
        Bt[(bq * 32 + j + 1) * 64 + bk] = (bf16)f.y;
        Bt[(bq * 32 + j + 2) * 64 + bk] = (bf16)f.z;
        Bt[(bq * 32 + j + 3) * 64 + bk] = (bf16)f.w;
      }
    }
    __syncthreads();
    const bf16* ar = Al + (wid * 16 + ln) * 64;
    #pragma unroll
    for (int kk = 0; kk < 64; kk += 32) {
      v16bf a = afrag(ar, kk, hi);
      #pragma unroll
      for (int h = 0; h < 2; ++h) {             // two batches of 4 N-tiles
        v16bf bfr[4];
        #pragma unroll
        for (int i = 0; i < 4; ++i)
          bfr[i] = *(const v16bf*)(Bt + ((h * 4 + i) * 16 + ln) * 64 + kk + hi * 16);
        #pragma unroll
        for (int i = 0; i < 4; ++i)
          acc[h * 4 + i] = wmma_bf16(a, bfr[i], acc[h * 4 + i]);
      }
    }
    __syncthreads();
  }

  #pragma unroll
  for (int nt = 0; nt < 8; ++nt) {
    int d = nt * 16 + ln;                       // 0..127: Q half then V half
    bf16* dstbuf = (d >= 64) ? Vb : Qb;
    int dl = d & 63;
    float bias = b1[cb + d];
    #pragma unroll
    for (int r = 0; r < 8; ++r) {
      int m = mbase + wid * 16 + r + hi * 8;
      int bb = m >> 11, s = m & 2047;
      dstbuf[((size_t)(bb * NH + hd) * SEQ + s) * HD + dl] = (bf16)(acc[nt][r] + bias);
    }
  }
}

// ---------------------------------------------------------------------------
// Kernel 2: flash attention per (b,head). grid=(SEQ/64, NH, NB), block=128.
// ---------------------------------------------------------------------------
__global__ __launch_bounds__(128) void attn(
    const bf16* __restrict__ Qb, const bf16* __restrict__ Vb, bf16* __restrict__ AO) {
  __shared__ __align__(32) bf16 Vs[32 * 64];     // V block row-major  [key][d]
  __shared__ __align__(32) bf16 VsT[64 * 32];    // V block transposed [d][key]
  __shared__ __align__(32) bf16 Ps[4 * 16 * 32]; // per-wave P scratch [m][k]

  const int t = threadIdx.x;
  const int wid = t >> 5, lane = t & 31, hi = lane >> 4, ln = lane & 15;
  const int b = blockIdx.z, hd = blockIdx.y;
  const int q0 = blockIdx.x * 64 + wid * 16;
  const size_t bh = (size_t)(b * NH + hd) * SEQ * HD;
  const bf16* Vp = Vb + bh;
  const bf16* qrow = Qb + bh + (size_t)(q0 + ln) * HD;

  v16bf aq0 = afrag(qrow, 0, hi);
  v16bf aq1 = afrag(qrow, 32, hi);

  float mr[8], lr[8];
  #pragma unroll
  for (int r = 0; r < 8; ++r) { mr[r] = -3.0e38f; lr[r] = 0.f; }
  v8f o[4] = {};
  bf16* Pw = Ps + wid * (16 * 32);
  const int vkey = t >> 2, vch = t & 3;
  const unsigned vs_lds = lds_off(Vs + vkey * 64 + vch * 16);

  for (int kb = 0; kb < SEQ / 32; ++kb) {
    {   // stage V block: async copy row-major; register path for transpose
      const bf16* src = Vp + (size_t)(kb * 32 + vkey) * HD + vch * 16;
      async_ld16<0>(vs_lds, (unsigned long long)src);      // Vs, ASYNCcnt
      async_ld16<16>(vs_lds, (unsigned long long)src);
      v8bf d0 = *(const v8bf*)(src);                       // L0/L2 hit
      v8bf d1 = *(const v8bf*)(src + 8);
      #pragma unroll
      for (int j = 0; j < 8; ++j) VsT[(vch * 16 + j) * 32 + vkey] = d0[j];
      #pragma unroll
      for (int j = 0; j < 8; ++j) VsT[(vch * 16 + 8 + j) * 32 + vkey] = d1[j];
    }
    wait_async0();
    __syncthreads();

    // scores S = Q(16x64) @ V^T(64x32)
    v8f s0 = {}, s1 = {};
    #pragma unroll
    for (int kk = 0; kk < 64; kk += 32) {
      v16bf a = (kk == 0) ? aq0 : aq1;
      v16bf b0  = *(const v16bf*)(Vs + (0  + ln) * 64 + kk + hi * 16);
      v16bf b1v = *(const v16bf*)(Vs + (16 + ln) * 64 + kk + hi * 16);
      s0 = wmma_bf16(a, b0, s0);
      s1 = wmma_bf16(a, b1v, s1);
    }

    // online softmax (C rows live across 16-lane groups)
    #pragma unroll
    for (int r = 0; r < 8; ++r) {
      float x0 = s0[r] * 0.125f, x1 = s1[r] * 0.125f;   // 1/sqrt(64)
      float tm = fmaxf(x0, x1);
      #pragma unroll
      for (int off = 1; off < 16; off <<= 1) tm = fmaxf(tm, __shfl_xor(tm, off, 16));
      float mnew  = fmaxf(mr[r], tm);
      float alpha = __expf(mr[r] - mnew);
      float p0 = __expf(x0 - mnew), p1 = __expf(x1 - mnew);
      float rs = p0 + p1;
      #pragma unroll
      for (int off = 1; off < 16; off <<= 1) rs += __shfl_xor(rs, off, 16);
      lr[r] = lr[r] * alpha + rs;
      mr[r] = mnew;
      o[0][r] *= alpha; o[1][r] *= alpha; o[2][r] *= alpha; o[3][r] *= alpha;
      Pw[(r + hi * 8) * 32 + ln]      = (bf16)p0;
      Pw[(r + hi * 8) * 32 + 16 + ln] = (bf16)p1;
    }
    // within-wave LDS RAW: P stores above -> A-frag loads below (DS in-order)
    asm volatile("s_wait_dscnt 0x0" ::: "memory");

    // O += P(16x32) @ V(32x64): batch all B-frag loads, then WMMA burst
    v16bf ap = afrag(Pw + ln * 32, 0, hi);
    v16bf bv[4];
    #pragma unroll
    for (int dt = 0; dt < 4; ++dt)
      bv[dt] = *(const v16bf*)(VsT + (dt * 16 + ln) * 32 + hi * 16);
    #pragma unroll
    for (int dt = 0; dt < 4; ++dt)
      o[dt] = wmma_bf16(ap, bv[dt], o[dt]);
    __syncthreads();
  }

  #pragma unroll
  for (int dt = 0; dt < 4; ++dt) {
    #pragma unroll
    for (int r = 0; r < 8; ++r) {
      int s = q0 + r + hi * 8;
      float val = o[dt][r] / lr[r];
      AO[((size_t)b * SEQ + s) * D_EMB + hd * HD + dt * 16 + ln] = (bf16)val;
    }
  }
}

// ---------------------------------------------------------------------------
// Kernel 3: output projection attn@W2+b2 -> f32. grid=(M/128, D/128), 256 thr.
// A tile is already bf16 -> pure async global->LDS copy.
// ---------------------------------------------------------------------------
__global__ __launch_bounds__(256) void out_proj(
    const bf16* __restrict__ AO, const float* __restrict__ W2,
    const float* __restrict__ b2, float* __restrict__ out) {
  __shared__ __align__(32) bf16 Al[128 * 64];
  __shared__ __align__(32) bf16 Bt[128 * 64];

  const int t = threadIdx.x;
  const int wid = t >> 5, lane = t & 31, hi = lane >> 4, ln = lane & 15;
  const int mbase = blockIdx.x * 128;
  const int cb = blockIdx.y * 128;
  const int arow = t >> 1, ahalf = t & 1;
  const int bk = t >> 2, bq = t & 3;
  const unsigned al_lds = lds_off(Al + arow * 64 + ahalf * 32);

  v8f acc[8] = {};

  for (int k0 = 0; k0 < D_EMB; k0 += 64) {
    {   // stage A: 64B/thread async copy straight into LDS
      unsigned long long src =
          (unsigned long long)(AO + (size_t)(mbase + arow) * D_EMB + k0 + ahalf * 32);
      async_ld16<0>(al_lds, src);
      async_ld16<16>(al_lds, src);
      async_ld16<32>(al_lds, src);
      async_ld16<48>(al_lds, src);
    }
    {   // stage B transposed from W2 (f32 -> bf16, manual)
      const float* src = W2 + (size_t)(k0 + bk) * D_EMB + cb + bq * 32;
      #pragma unroll
      for (int j = 0; j < 32; j += 4) {
        float4 f = *(const float4*)(src + j);
        Bt[(bq * 32 + j + 0) * 64 + bk] = (bf16)f.x;
        Bt[(bq * 32 + j + 1) * 64 + bk] = (bf16)f.y;
        Bt[(bq * 32 + j + 2) * 64 + bk] = (bf16)f.z;
        Bt[(bq * 32 + j + 3) * 64 + bk] = (bf16)f.w;
      }
    }
    wait_async0();
    __syncthreads();
    const bf16* ar = Al + (wid * 16 + ln) * 64;
    #pragma unroll
    for (int kk = 0; kk < 64; kk += 32) {
      v16bf a = afrag(ar, kk, hi);
      #pragma unroll
      for (int h = 0; h < 2; ++h) {
        v16bf bfr[4];
        #pragma unroll
        for (int i = 0; i < 4; ++i)
          bfr[i] = *(const v16bf*)(Bt + ((h * 4 + i) * 16 + ln) * 64 + kk + hi * 16);
        #pragma unroll
        for (int i = 0; i < 4; ++i)
          acc[h * 4 + i] = wmma_bf16(a, bfr[i], acc[h * 4 + i]);
      }
    }
    __syncthreads();
  }

  #pragma unroll
  for (int nt = 0; nt < 8; ++nt) {
    float bias = b2[cb + nt * 16 + ln];
    #pragma unroll
    for (int r = 0; r < 8; ++r) {
      int m = mbase + wid * 16 + r + hi * 8;
      out[(size_t)m * D_EMB + cb + nt * 16 + ln] = acc[nt][r] + bias;
    }
  }
}

// ---------------------------------------------------------------------------
extern "C" void kernel_launch(void* const* d_in, const int* in_sizes, int n_in,
                              void* d_out, int out_size, void* d_ws, size_t ws_size,
                              hipStream_t stream) {
  (void)in_sizes; (void)n_in; (void)out_size; (void)ws_size;
  const float* x  = (const float*)d_in[0];
  const float* W1 = (const float*)d_in[1];
  const float* b1 = (const float*)d_in[2];
  const float* W2 = (const float*)d_in[3];
  const float* b2 = (const float*)d_in[4];
  float* out = (float*)d_out;

  constexpr size_t QV = (size_t)NB * NH * SEQ * HD;   // 6,291,456 elems
  bf16* Qb = (bf16*)d_ws;
  bf16* Vb = Qb + QV;
  bf16* AO = Vb + QV;                                  // [B,S,D] bf16
  // workspace: 3*QV*2 bytes ~= 37.7 MB (L2-resident @ 192 MB)

  qkv_gemm<<<dim3((NB * SEQ) / 128, NH),          256, 0, stream>>>(x, W1, b1, Qb, Vb);
  attn    <<<dim3(SEQ / 64, NH, NB),              128, 0, stream>>>(Qb, Vb, AO);
  out_proj<<<dim3((NB * SEQ) / 128, D_EMB / 128), 256, 0, stream>>>(AO, W2, b2, out);
}
